// HypergraphContrastiveLoss_62792421867973
// MI455X (gfx1250) — compile-verified, matrix operand order checked by reading, not run on hardware
//
#include <hip/hip_runtime.h>
#include <math.h>

// ---------------------------------------------------------------------------
// HypergraphContrastiveLoss on MI455X (gfx1250)
// loss = mean_i( log( sum_j exp(<s_i, t_j>/T) ) - <s_i, t_i>/T )
// (diagonal mask + duplicated positive column cancel exactly, so no masking)
//
// Core: 9216x9216x256 GEMM via v_wmma_f32_16x16x32_f16 with fused exp+rowsum.
// B panels are staged through LDS with the Tensor Data Mover
// (tensor_load_to_lds, double-buffered, s_wait_tensorcnt) so the 8 waves of a
// block share each 16x256 f16 tile: 8x less L2 read traffic, and the TDM copy
// of tile t+1 overlaps the WMMA work on tile t.
// ---------------------------------------------------------------------------

typedef _Float16 half8  __attribute__((ext_vector_type(8)));
typedef _Float16 half16 __attribute__((ext_vector_type(16)));
typedef float    v8f    __attribute__((ext_vector_type(8)));
typedef unsigned int u32x4 __attribute__((ext_vector_type(4)));
typedef int          i32x4 __attribute__((ext_vector_type(4)));
typedef int          i32x8 __attribute__((ext_vector_type(8)));

#define BN_TOT   9216                   // B*N = 16*576
#define CDIM     256
#define TEMP_INV 14.2857142857142857f   // 1/0.07

#define ROWS_PER_BLOCK 128              // 8 waves x 16-row stripe
#define NSPLIT   4                      // column-space splits (occupancy)
#define NCOLT    (BN_TOT / 16)          // 576 column tiles
#define CT_PER_SPLIT (NCOLT / NSPLIT)   // 144

// LDS tile: 16 rows x 256 halves, padded by TDM to 264 halves (528 B) per row
// -> per-lane ds_load_b128 addresses land 4 banks apart (conflict-free).
#define LDS_ROW_H   264
#define LDS_TILE_B  (16 * LDS_ROW_H * 2)   // 8448 bytes
#define LDS_TOT_I32 (2 * 16 * LDS_ROW_H / 2)  // total smem in dwords (4224)

union FragAB { half16 h16; half8 h8[2]; };

// ---------------------------------------------------------------------------
// TDM: issue a 16x256 f16 tile load (row-major, stride 256) into LDS.
// D# fields per CDNA5 ISA:
//  g0: count=1 | lds_addr | global_addr(57b) | type=2
//  g1: data_size=2B, pad_enable, pad_interval=128 DW (512B), pad_amount=4 DW,
//      tensor_dim0=256, tensor_dim1=16, tile_dim0=256, tile_dim1=16,
//      tensor_dim0_stride=256
// ---------------------------------------------------------------------------
__device__ __forceinline__ void tdm_load_tile16x256(unsigned ldsOff,
                                                    const _Float16* gptr)
{
    unsigned long long ga = (unsigned long long)(size_t)gptr;
    u32x4 g0;
    g0[0] = 1u;                                        // count=1, user load
    g0[1] = ldsOff;                                    // lds_addr (bytes)
    g0[2] = (unsigned)ga;                              // global_addr[31:0]
    g0[3] = (unsigned)((ga >> 32) & 0x01FFFFFFu)       // global_addr[56:32]
          | 0x80000000u;                               // type=2 ("image")
    i32x8 g1;
    g1[0] = 0x07910000;      // data_size=1(2B)<<16 | pad_en<<20 | intvl=6<<22 | amt=3<<25
    g1[1] = 0x01000000;      // tensor_dim0=256 -> bits[79:48] low16 in [31:16]
    g1[2] = 0x00100000;      // tensor_dim1=16  -> bits[111:80] low16 in [31:16]
    g1[3] = 0x01000000;      // tile_dim0=256   -> bits[127:112]
    g1[4] = 16;              // tile_dim1=16, tile_dim2=0
    g1[5] = 256;             // tensor_dim0_stride low32
    g1[6] = 0;
    g1[7] = 0;
    i32x4 g2 = {0, 0, 0, 0};
    i32x4 g3 = {0, 0, 0, 0};
#if defined(__clang_major__) && __clang_major__ >= 23
    i32x8 g4 = {0, 0, 0, 0, 0, 0, 0, 0};
    __builtin_amdgcn_tensor_load_to_lds(g0, g1, g2, g3, g4, 0);
#else
    __builtin_amdgcn_tensor_load_to_lds(g0, g1, g2, g3, 0);
#endif
}

// ---------------------------------------------------------------------------
// Kernel 1: per-row L2 normalize, convert to f16, diagonal dot, zero rowsums.
// ---------------------------------------------------------------------------
__global__ __launch_bounds__(256)
void ncl_normalize_kernel(const float* __restrict__ src,
                          const float* __restrict__ tgt,
                          _Float16* __restrict__ srcH,
                          _Float16* __restrict__ tgtH,
                          float* __restrict__ dpos,
                          float* __restrict__ rowsum)
{
    const int row = blockIdx.x;
    const int c   = threadIdx.x;
    const size_t idx = (size_t)row * CDIM + c;

    const float s = src[idx];
    const float t = tgt[idx];
    float ss = s * s, tt = t * t, st = s * t;

    #pragma unroll
    for (int m = 16; m >= 1; m >>= 1) {
        ss += __shfl_xor(ss, m, 32);
        tt += __shfl_xor(tt, m, 32);
        st += __shfl_xor(st, m, 32);
    }
    __shared__ float red[8][3];
    const int wave = threadIdx.x >> 5;
    const int lane = threadIdx.x & 31;
    if (lane == 0) { red[wave][0] = ss; red[wave][1] = tt; red[wave][2] = st; }
    __syncthreads();
    ss = tt = st = 0.0f;
    #pragma unroll
    for (int w = 0; w < 8; ++w) { ss += red[w][0]; tt += red[w][1]; st += red[w][2]; }

    const float ns = fmaxf(sqrtf(ss), 1e-12f);   // matches F.normalize eps
    const float nt = fmaxf(sqrtf(tt), 1e-12f);

    srcH[idx] = (_Float16)(s / ns);
    tgtH[idx] = (_Float16)(t / nt);

    if (threadIdx.x == 0) {
        dpos[row]   = st / (ns * nt);            // positive similarity (fp32 exact)
        rowsum[row] = 0.0f;                      // init for atomic accumulation
    }
}

// ---------------------------------------------------------------------------
// Kernel 2: TDM-staged, LDS-shared WMMA GEMM with fused exp + row-sum.
//
// blockIdx = rowGroup * NSPLIT + colSplit.
// Wave w owns rows rowGroup*128 + w*16 .. +15 (A frags in registers, K=256).
// All 8 waves share each B tile (16 cols x 256 K, f16) from double-buffered
// LDS, filled by wave 0 via tensor_load_to_lds one tile ahead.
//
// smem is zero-filled with visible LDS stores first so the compiler can never
// treat the (TDM-written) buffer as undef memory -> real ds_load_b128 per
// fragment; __syncthreads fences block forwarding of the zeros.
//
// All 8 B fragments are loaded into distinct registers before the WMMA chain:
// a 16-wide ds_load_b128 clause, one dscnt wait, then 8 back-to-back WMMAs.
//
// f16 fragment layout (16x32 step): lane r=lane&15 -> row; kb=(lane>=16)?8:0;
// VGPR0..3 = K kb..kb+7, VGPR4..7 = K kb+16..kb+23 -> two b128 loads.
// D layout: lanes 0-15 hold M=0..7 (vgpr=M), lanes 16-31 hold M=8..15.
// ---------------------------------------------------------------------------
__global__ __launch_bounds__(256, 1)
void ncl_simexp_kernel(const _Float16* __restrict__ srcH,
                       const _Float16* __restrict__ tgtH,
                       float* __restrict__ rowsum)
{
    __shared__ _Float16 smem[2][16][LDS_ROW_H];   // only LDS object -> offset 0

    const int rowGrp   = blockIdx.x / NSPLIT;
    const int colSplit = blockIdx.x % NSPLIT;
    const int ctBegin  = colSplit * CT_PER_SPLIT;

    const int wave = threadIdx.x >> 5;
    const int lane = threadIdx.x & 31;
    const int r    = lane & 15;
    const int kb   = (lane >> 4) << 3;            // 0 or 8

    const int waveRowBase = rowGrp * ROWS_PER_BLOCK + wave * 16;

    // Visible initialization of smem (overwritten by TDM before first use).
    {
        int* sm32 = (int*)&smem[0][0][0];
        for (int j = threadIdx.x; j < LDS_TOT_I32; j += 256)
            sm32[j] = 0;
    }
    __syncthreads();                              // init done before TDM issue

    // Preload this wave's A fragments (16 rows x K=256).
    FragAB a[8];
    const _Float16* aRow = srcH + (size_t)(waveRowBase + r) * CDIM + kb;
    #pragma unroll
    for (int k0 = 0; k0 < 8; ++k0) {
        a[k0].h8[0] = *reinterpret_cast<const half8*>(aRow + k0 * 32);
        a[k0].h8[1] = *reinterpret_cast<const half8*>(aRow + k0 * 32 + 16);
    }

    float racc[8];
    #pragma unroll
    for (int v = 0; v < 8; ++v) racc[v] = 0.0f;

    // Prologue: wave 0 kicks TDM for the first tile into buffer 0.
    if (threadIdx.x < 32)
        tdm_load_tile16x256(0u, tgtH + (size_t)ctBegin * 16 * CDIM);

    for (int i = 0; i < CT_PER_SPLIT; ++i) {
        const int cur = i & 1;
        if (threadIdx.x < 32) {
            if (i + 1 < CT_PER_SPLIT) {
                // Issue next tile into the other buffer, then wait only for
                // the older (in-order) TDM op: copy of tile i+1 overlaps WMMA.
                tdm_load_tile16x256(cur ? 0u : (unsigned)LDS_TILE_B,
                                    tgtH + (size_t)(ctBegin + i + 1) * 16 * CDIM);
                __builtin_amdgcn_s_wait_tensorcnt(1);
            } else {
                __builtin_amdgcn_s_wait_tensorcnt(0);
            }
        }
        __syncthreads();                          // buf[cur] valid for everyone
        asm volatile("" ::: "memory");            // TDM wrote LDS behind compiler's back

        const _Float16* bRow = &smem[cur][r][kb]; // stays addrspace(3) -> ds_load

        // Batch-load all 8 B fragments (16 x ds_load_b128, distinct registers).
        FragAB b[8];
        #pragma unroll
        for (int k0 = 0; k0 < 8; ++k0) {
            b[k0].h8[0] = *reinterpret_cast<const half8*>(bRow + k0 * 32);
            b[k0].h8[1] = *reinterpret_cast<const half8*>(bRow + k0 * 32 + 16);
        }

        v8f acc = {};
        #pragma unroll
        for (int k0 = 0; k0 < 8; ++k0)
            acc = __builtin_amdgcn_wmma_f32_16x16x32_f16(
                      false, a[k0].h16, false, b[k0].h16,
                      (short)0, acc, false, false);

        #pragma unroll
        for (int v = 0; v < 8; ++v)
            racc[v] += __expf(acc[v] * TEMP_INV);

        __syncthreads();                          // all waves done with buf[cur]
    }

    // Reduce across the 16 lanes of each half (N dimension of the tile).
    #pragma unroll
    for (int v = 0; v < 8; ++v) {
        float x = racc[v];
        x += __shfl_xor(x, 1, 32);
        x += __shfl_xor(x, 2, 32);
        x += __shfl_xor(x, 4, 32);
        x += __shfl_xor(x, 8, 32);
        racc[v] = x;
    }
    if ((lane & 15) == 0) {
        const int mBase = (lane >> 4) * 8;        // lanes 0-15 -> M 0..7; 16-31 -> 8..15
        #pragma unroll
        for (int v = 0; v < 8; ++v)
            atomicAdd(&rowsum[waveRowBase + mBase + v], racc[v]);  // global_atomic_add_f32
    }
}

// ---------------------------------------------------------------------------
// Kernel 3: loss = mean_i( log(rowsum[i]) - dpos[i]/T )
// ---------------------------------------------------------------------------
__global__ __launch_bounds__(256)
void ncl_loss_kernel(const float* __restrict__ rowsum,
                     const float* __restrict__ dpos,
                     float* __restrict__ out)
{
    float p = 0.0f;
    for (int i = threadIdx.x; i < BN_TOT; i += 256)
        p += logf(rowsum[i]) - dpos[i] * TEMP_INV;

    #pragma unroll
    for (int m = 16; m >= 1; m >>= 1) p += __shfl_xor(p, m, 32);

    __shared__ float red[8];
    const int wave = threadIdx.x >> 5;
    const int lane = threadIdx.x & 31;
    if (lane == 0) red[wave] = p;
    __syncthreads();
    if (threadIdx.x == 0) {
        float tot = 0.0f;
        #pragma unroll
        for (int w = 0; w < 8; ++w) tot += red[w];
        out[0] = tot / (float)BN_TOT;
    }
}

// ---------------------------------------------------------------------------
extern "C" void kernel_launch(void* const* d_in, const int* in_sizes, int n_in,
                              void* d_out, int out_size, void* d_ws, size_t ws_size,
                              hipStream_t stream)
{
    (void)in_sizes; (void)n_in; (void)out_size; (void)ws_size;

    const float* src = (const float*)d_in[0];    // [16,576,256] f32
    const float* tgt = (const float*)d_in[1];

    // workspace: srcH (4.7MB f16) | tgtH (4.7MB f16) | dpos (36KB) | rowsum (36KB)
    char* ws = (char*)d_ws;
    _Float16* srcH   = (_Float16*)ws;
    _Float16* tgtH   = srcH + (size_t)BN_TOT * CDIM;
    float*    dpos   = (float*)(tgtH + (size_t)BN_TOT * CDIM);
    float*    rowsum = dpos + BN_TOT;
    float*    out    = (float*)d_out;

    ncl_normalize_kernel<<<BN_TOT, 256, 0, stream>>>(src, tgt, srcH, tgtH, dpos, rowsum);
    ncl_simexp_kernel<<<(BN_TOT / ROWS_PER_BLOCK) * NSPLIT, 256, 0, stream>>>(srcH, tgtH, rowsum);
    ncl_loss_kernel<<<1, 256, 0, stream>>>(rowsum, dpos, out);
}
